// Share_Attention_25735444038357
// MI455X (gfx1250) — matrix operand order repaired and checked by reading, not compiled, not run on hardware
//
#include <hip/hip_runtime.h>

typedef __attribute__((ext_vector_type(16))) _Float16 v16h;
typedef __attribute__((ext_vector_type(8)))  _Float16 v8h;
typedef __attribute__((ext_vector_type(8)))  float    v8f;

#define BB 4
#define NSEQ 2048
#define DIM 512
#define INNER 512
#define HPB 4              // heads per branch (ALPHA = 0.5)
#define DHEAD 128          // INNER / HPB
#define ROWS (BB*NSEQ)     // 8192
#define QKV_COLS (3*INNER) // 1536
#define ATT_SCALE 0.125f   // DIM_HEAD(64)^-0.5 per reference

union AU { v16h v; _Float16 h[16]; };
union CU { v8f  v; float    f[8];  };
union H8 { v8h  v; _Float16 h[8];  };

__device__ __forceinline__ v8f wmma_f16(const AU& a, const AU& b, v8f c) {
    return __builtin_amdgcn_wmma_f32_16x16x32_f16(false, a.v, false, b.v,
                                                  (short)0, c, false, false);
}

// Async copy 16 bytes global -> LDS (CDNA5 async path, ASYNCcnt-tracked)
__device__ __forceinline__ void async_b128(uint32_t lds_off, const void* gptr) {
    asm volatile("global_load_async_to_lds_b128 %0, %1, off"
                 :: "v"(lds_off), "v"((uint64_t)(uintptr_t)gptr) : "memory");
}
__device__ __forceinline__ void wait_async0() {
    asm volatile("s_wait_asynccnt 0x0" ::: "memory");
}
__device__ __forceinline__ uint32_t lds_off32(const void* p) {
    return (uint32_t)(uintptr_t)p;   // addrspace(3) offset lives in the low 32 bits
}

// ---------------------------------------------------------------------------
// Kernel 0: cast x (f32) -> f16 once.  Each x row feeds 24 column-blocks of the
// QKV GEMM, so converting up front removes 24x redundant VALU cvt work.
// ---------------------------------------------------------------------------
__global__ __launch_bounds__(256)
void cast_x_kernel(const float* __restrict__ x1, const float* __restrict__ x2,
                   _Float16* __restrict__ xbuf)
{
    const float* __restrict__ X = blockIdx.y ? x2 : x1;
    const size_t base = ((size_t)blockIdx.x * 256 + threadIdx.x) * 8;
    const float4 f0 = *(const float4*)(X + base);
    const float4 f1 = *(const float4*)(X + base + 4);
    H8 o;
    o.h[0] = (_Float16)f0.x; o.h[1] = (_Float16)f0.y;
    o.h[2] = (_Float16)f0.z; o.h[3] = (_Float16)f0.w;
    o.h[4] = (_Float16)f1.x; o.h[5] = (_Float16)f1.y;
    o.h[6] = (_Float16)f1.z; o.h[7] = (_Float16)f1.w;
    *(v8h*)(xbuf + (size_t)blockIdx.y * ROWS * DIM + base) = o.v;
}

// ---------------------------------------------------------------------------
// Shared helper: async-stage a 32x64 f32 weight chunk, then convert it ONCE to
// f16 transposed [n][k] in LDS so B-operand builds are two ds_load_b128 each.
// ---------------------------------------------------------------------------
__device__ __forceinline__
void stage_w_chunk(const float* __restrict__ Wsrc, int ld, int k0, int colBase,
                   float* lWf, _Float16* lWh, int tid)
{
    const int krow  = tid >> 2;          // 0..31
    const int cpart = (tid & 3) * 16;    // 0,16,32,48
    {
        const float* src = Wsrc + (size_t)(k0 + krow) * ld + colBase + cpart;
        const uint32_t dst = lds_off32(lWf + krow * 64 + cpart);
        #pragma unroll
        for (int u = 0; u < 4; ++u) async_b128(dst + u * 16, src + u * 4);
    }
    wait_async0();
    __syncthreads();                     // f32 chunk visible; prev-iter lWh reads fenced
    {
        const int n  = tid >> 1;         // 0..63
        const int kh = (tid & 1) * 16;   // 0,16
        _Float16* dst = lWh + n * 32 + kh;
        #pragma unroll
        for (int e = 0; e < 16; ++e)
            dst[e] = (_Float16)lWf[(size_t)(kh + e) * 64 + n];
    }
    __syncthreads();                     // f16 transposed chunk visible
}

__device__ __forceinline__ AU load_b_op(const _Float16* lWh, int nt, int lid, int half) {
    AU b;
    const _Float16* wrow = lWh + (nt * 16 + lid) * 32 + half * 16;
    H8 x0 = *(const H8*)wrow, x1 = *(const H8*)(wrow + 8);
    #pragma unroll
    for (int e = 0; e < 8; ++e) { b.h[e] = x0.h[e]; b.h[8 + e] = x1.h[e]; }
    return b;
}

// ---------------------------------------------------------------------------
// Kernel 1: qkv = x @ W_qkv  (f16 WMMA, f32 accum), output scattered into
// q [bh][n][d], k [bh][n][d], v TRANSPOSED [bh][d][n], all f16.
// ---------------------------------------------------------------------------
__global__ __launch_bounds__(128)
void qkv_gemm_kernel(const _Float16* __restrict__ xbuf, const float* __restrict__ Wqkv,
                     _Float16* __restrict__ qbuf, _Float16* __restrict__ kbuf,
                     _Float16* __restrict__ vbuf)
{
    __shared__ float    lWf[32 * 64];   // staged f32 chunk [k][n], 8 KB
    __shared__ _Float16 lWh[64 * 32];   // converted f16 chunk [n][k], 4 KB

    const int branch = blockIdx.z;
    const int tid  = threadIdx.x;
    const int wave = tid >> 5;
    const int lane = tid & 31;
    const int lid  = lane & 15;
    const int half = lane >> 4;
    const int rowBase = blockIdx.y * 64 + wave * 16;
    const int colBase = blockIdx.x * 64;

    v8f acc[4];
    #pragma unroll
    for (int i = 0; i < 4; ++i) acc[i] = v8f{0.f,0.f,0.f,0.f,0.f,0.f,0.f,0.f};

    const _Float16* xrow = xbuf + ((size_t)branch * ROWS + rowBase + lid) * DIM;

    for (int k0 = 0; k0 < DIM; k0 += 32) {
        if (k0 + 32 < DIM)
            __builtin_prefetch(Wqkv + (size_t)(k0 + 32 + (tid >> 2)) * QKV_COLS
                                    + colBase + (tid & 3) * 16, 0, 3);
        stage_w_chunk(Wqkv, QKV_COLS, k0, colBase, lWf, lWh, tid);

        AU a;
        H8 r1 = *(const H8*)(xrow + k0 + half * 16);
        H8 r2 = *(const H8*)(xrow + k0 + half * 16 + 8);
        #pragma unroll
        for (int e = 0; e < 8; ++e) { a.h[e] = r1.h[e]; a.h[8 + e] = r2.h[e]; }

        #pragma unroll
        for (int nt = 0; nt < 4; ++nt)
            acc[nt] = wmma_f16(a, load_b_op(lWh, nt, lid, half), acc[nt]);
    }

    #pragma unroll
    for (int nt = 0; nt < 4; ++nt) {
        const int col   = colBase + nt * 16 + lid;
        const int which = col / INNER;
        const int ic    = col % INNER;
        const int head  = ic / DHEAD;
        const int dc    = ic % DHEAD;
        CU c; c.v = acc[nt];
        #pragma unroll
        for (int r = 0; r < 8; ++r) {
            const int grow  = rowBase + r + half * 8;
            const int batch = grow >> 11;          // /2048
            const int nn    = grow & (NSEQ - 1);
            const size_t bh = (size_t)(branch * BB + batch) * HPB + head;
            const _Float16 val = (_Float16)c.f[r];
            if (which == 0)
                qbuf[bh * (size_t)(NSEQ * DHEAD) + (size_t)nn * DHEAD + dc] = val;
            else if (which == 1)
                kbuf[bh * (size_t)(NSEQ * DHEAD) + (size_t)nn * DHEAD + dc] = val;
            else
                vbuf[bh * (size_t)(DHEAD * NSEQ) + (size_t)dc * NSEQ + nn] = val;
        }
    }
}

// ---------------------------------------------------------------------------
// Kernel 2: flash attention per (branch, b*h).  Block = 4 waves; each wave owns
// a 16-row query tile; K/V tiles (32 keys) async-staged into LDS, shared by all
// waves.
// ---------------------------------------------------------------------------
__global__ __launch_bounds__(128)
void attn_kernel(const _Float16* __restrict__ qbuf, const _Float16* __restrict__ kbuf,
                 const _Float16* __restrict__ vbuf, _Float16* __restrict__ abuf)
{
    __shared__ _Float16 lK[32 * 128];    // [key][feat]
    __shared__ _Float16 lVt[128 * 32];   // [feat][key]
    __shared__ _Float16 lP[4][16 * 32];  // per-wave P transpose buffer

    const int branch = blockIdx.z;
    const int bh     = blockIdx.y;       // 0..15
    const int batch  = bh >> 2;
    const int head   = bh & 3;
    const int tid  = threadIdx.x;
    const int wave = tid >> 5;
    const int lane = tid & 31;
    const int lid  = lane & 15;
    const int half = lane >> 4;

    const size_t bhIdx = (size_t)(branch * BB + batch) * HPB + head;
    const _Float16* qb = qbuf + bhIdx * (size_t)(NSEQ * DHEAD);
    const _Float16* kb = kbuf + bhIdx * (size_t)(NSEQ * DHEAD);
    const _Float16* vb = vbuf + bhIdx * (size_t)(DHEAD * NSEQ);

    const int q0 = blockIdx.x * 64 + wave * 16;

    // Q tile as 4 A-operands (K-chunks of 32 over d=128), kept in VGPRs.
    AU aq[4];
    {
        const _Float16* qrow = qb + (size_t)(q0 + lid) * DHEAD;
        #pragma unroll
        for (int kc = 0; kc < 4; ++kc) {
            H8 r1 = *(const H8*)(qrow + kc * 32 + half * 16);
            H8 r2 = *(const H8*)(qrow + kc * 32 + half * 16 + 8);
            #pragma unroll
            for (int e = 0; e < 8; ++e) { aq[kc].h[e] = r1.h[e]; aq[kc].h[8 + e] = r2.h[e]; }
        }
    }

    float m[8], l[8];
    v8f o[8];
    #pragma unroll
    for (int r = 0; r < 8; ++r) { m[r] = -1e30f; l[r] = 0.f; }
    #pragma unroll
    for (int c = 0; c < 8; ++c) o[c] = v8f{0.f,0.f,0.f,0.f,0.f,0.f,0.f,0.f};

    const int key  = tid >> 2;        // 0..31
    const int part = (tid & 3) * 32;  // 0,32,64,96 (halves)

    for (int j0 = 0; j0 < NSEQ; j0 += 32) {
        __syncthreads();
        {   // async-stage K tile: 32 keys x 128 feats (row-major per key)
            const _Float16* src = kb + (size_t)(j0 + key) * DHEAD + part;
            const uint32_t dst = lds_off32(lK + key * 128 + part);
            #pragma unroll
            for (int u = 0; u < 4; ++u) async_b128(dst + u * 16, src + u * 8);
        }
        {   // async-stage V^T tile: 128 feats x 32 keys
            const _Float16* src = vb + (size_t)tid * NSEQ + j0;
            const uint32_t dst = lds_off32(lVt + tid * 32);
            #pragma unroll
            for (int u = 0; u < 4; ++u) async_b128(dst + u * 16, src + u * 8);
        }
        wait_async0();
        __syncthreads();

        // S = Q K^T : two 16x16 accumulators (keys 0-15, 16-31), K-dim = 128
        v8f s0 = v8f{0.f,0.f,0.f,0.f,0.f,0.f,0.f,0.f};
        v8f s1 = s0;
        #pragma unroll
        for (int kc = 0; kc < 4; ++kc) {
            AU b0, b1;
            const _Float16* kr0 = lK + lid * 128 + kc * 32 + half * 16;
            const _Float16* kr1 = lK + (16 + lid) * 128 + kc * 32 + half * 16;
            H8 x0 = *(const H8*)kr0, x1 = *(const H8*)(kr0 + 8);
            H8 y0 = *(const H8*)kr1, y1 = *(const H8*)(kr1 + 8);
            #pragma unroll
            for (int e = 0; e < 8; ++e) {
                b0.h[e] = x0.h[e]; b0.h[8 + e] = x1.h[e];
                b1.h[e] = y0.h[e]; b1.h[8 + e] = y1.h[e];
            }
            s0 = wmma_f16(aq[kc], b0, s0);
            s1 = wmma_f16(aq[kc], b1, s1);
        }

        // Online softmax (row = r + 8*half; N spans the 16-lane group)
        CU cs0, cs1; cs0.v = s0; cs1.v = s1;
        _Float16* Pw = lP[wave];
        float corr[8];
        #pragma unroll
        for (int r = 0; r < 8; ++r) {
            float a0 = cs0.f[r] * ATT_SCALE;
            float a1 = cs1.f[r] * ATT_SCALE;
            float mx = fmaxf(a0, a1);
            #pragma unroll
            for (int off = 1; off < 16; off <<= 1) mx = fmaxf(mx, __shfl_xor(mx, off, 16));
            const float mnew = fmaxf(m[r], mx);
            corr[r] = __expf(m[r] - mnew);
            const float p0 = __expf(a0 - mnew);
            const float p1 = __expf(a1 - mnew);
            float rs = p0 + p1;
            #pragma unroll
            for (int off = 1; off < 16; off <<= 1) rs += __shfl_xor(rs, off, 16);
            l[r] = l[r] * corr[r] + rs;
            m[r] = mnew;
            Pw[(r + half * 8) * 32 + lid]      = (_Float16)p0;
            Pw[(r + half * 8) * 32 + 16 + lid] = (_Float16)p1;
        }
        #pragma unroll
        for (int c = 0; c < 8; ++c) {
            CU t; t.v = o[c];
            #pragma unroll
            for (int r = 0; r < 8; ++r) t.f[r] *= corr[r];
            o[c] = t.v;
        }
        __builtin_amdgcn_wave_barrier();   // per-wave LDS transpose: keep store->load order

        // P (16x32, A-layout) read back from LDS
        AU ap;
        {
            const _Float16* prow = Pw + lid * 32;
            H8 r1 = *(const H8*)(prow + half * 16);
            H8 r2 = *(const H8*)(prow + half * 16 + 8);
            #pragma unroll
            for (int e = 0; e < 8; ++e) { ap.h[e] = r1.h[e]; ap.h[8 + e] = r2.h[e]; }
        }

        // O += P @ V : 8 N-chunks of the 128-wide head dim, K-dim = 32 keys
        #pragma unroll
        for (int c = 0; c < 8; ++c) {
            AU bv;
            const _Float16* vrow = lVt + (c * 16 + lid) * 32 + half * 16;
            H8 x0 = *(const H8*)vrow, x1 = *(const H8*)(vrow + 8);
            #pragma unroll
            for (int e = 0; e < 8; ++e) { bv.h[e] = x0.h[e]; bv.h[8 + e] = x1.h[e]; }
            o[c] = wmma_f16(ap, bv, o[c]);
        }
    }

    // Normalize and write attention output [branch][b][n][inner] (f16)
    float inv[8];
    #pragma unroll
    for (int r = 0; r < 8; ++r) inv[r] = 1.f / l[r];
    _Float16* arow = abuf + (size_t)branch * ROWS * INNER;
    #pragma unroll
    for (int c = 0; c < 8; ++c) {
        CU t; t.v = o[c];
        const int col = head * DHEAD + c * 16 + lid;
        #pragma unroll
        for (int r = 0; r < 8; ++r) {
            const int nn = q0 + r + half * 8;
            arow[((size_t)batch * NSEQ + nn) * INNER + col] = (_Float16)(t.f[r] * inv[r]);
        }
    }
}

// ---------------------------------------------------------------------------
// Kernel 3: out = attn @ W_out + b_out  (f16 WMMA, f32 accum, f32 out)
// ---------------------------------------------------------------------------
__global__ __launch_bounds__(128)
void out_gemm_kernel(const _Float16* __restrict__ abuf, const float* __restrict__ Wout,
                     const float* __restrict__ bout, float* __restrict__ out)
{
    __shared__ float    lWf[32 * 64];
    __shared__ _Float16 lWh[64 * 32];

    const int branch = blockIdx.z;
    const int tid  = threadIdx.x;
    const int wave = tid >> 5;
    const int lane = tid & 31;
    const int lid  = lane & 15;
    const int half = lane >> 4;
    const int rowBase = blockIdx.y * 64 + wave * 16;
    const int colBase = blockIdx.x * 64;

    v8f acc[4];
    #pragma unroll
    for (int i = 0; i < 4; ++i) acc[i] = v8f{0.f,0.f,0.f,0.f,0.f,0.f,0.f,0.f};

    const _Float16* arow = abuf + ((size_t)branch * ROWS + rowBase + lid) * INNER;

    for (int k0 = 0; k0 < INNER; k0 += 32) {
        if (k0 + 32 < INNER)
            __builtin_prefetch(Wout + (size_t)(k0 + 32 + (tid >> 2)) * DIM
                                    + colBase + (tid & 3) * 16, 0, 3);
        stage_w_chunk(Wout, DIM, k0, colBase, lWf, lWh, tid);

        AU a;
        H8 r1 = *(const H8*)(arow + k0 + half * 16);
        H8 r2 = *(const H8*)(arow + k0 + half * 16 + 8);
        #pragma unroll
        for (int e = 0; e < 8; ++e) { a.h[e] = r1.h[e]; a.h[8 + e] = r2.h[e]; }

        #pragma unroll
        for (int nt = 0; nt < 4; ++nt)
            acc[nt] = wmma_f16(a, load_b_op(lWh, nt, lid, half), acc[nt]);
    }

    #pragma unroll
    for (int nt = 0; nt < 4; ++nt) {
        const int col = colBase + nt * 16 + lid;
        const float bias = bout[col];
        CU c; c.v = acc[nt];
        #pragma unroll
        for (int r = 0; r < 8; ++r) {
            const int grow = rowBase + r + half * 8;
            out[(size_t)branch * ROWS * DIM + (size_t)grow * DIM + col] = c.f[r] + bias;
        }
    }
}

// ---------------------------------------------------------------------------
extern "C" void kernel_launch(void* const* d_in, const int* in_sizes, int n_in,
                              void* d_out, int out_size, void* d_ws, size_t ws_size,
                              hipStream_t stream)
{
    (void)in_sizes; (void)n_in; (void)out_size; (void)ws_size;
    const float* x1   = (const float*)d_in[0];
    const float* x2   = (const float*)d_in[1];
    const float* Wqkv = (const float*)d_in[2];
    const float* Wout = (const float*)d_in[3];
    const float* bout = (const float*)d_in[4];
    float* out = (float*)d_out;

    _Float16* ws = (_Float16*)d_ws;
    const size_t XSZ = (size_t)2 * ROWS * DIM;               // 8,388,608 halves
    const size_t QSZ = (size_t)2 * BB * HPB * NSEQ * DHEAD;  // 8,388,608 halves each
    _Float16* xbuf = ws;
    _Float16* qbuf = xbuf + XSZ;
    _Float16* kbuf = qbuf + QSZ;
    _Float16* vbuf = kbuf + QSZ;
    _Float16* abuf = vbuf + QSZ;

    cast_x_kernel<<<dim3((ROWS * DIM) / (256 * 8), 2, 1), dim3(256, 1, 1), 0, stream>>>(
        x1, x2, xbuf);
    dim3 blk(128, 1, 1);
    qkv_gemm_kernel<<<dim3(QKV_COLS / 64, ROWS / 64, 2), blk, 0, stream>>>(
        xbuf, Wqkv, qbuf, kbuf, vbuf);
    attn_kernel<<<dim3(NSEQ / 64, BB * HPB, 2), blk, 0, stream>>>(
        qbuf, kbuf, vbuf, abuf);
    out_gemm_kernel<<<dim3(DIM / 64, ROWS / 64, 2), blk, 0, stream>>>(
        abuf, Wout, bout, out);
}
